// MultiHeadAttention_68719476736302
// MI455X (gfx1250) — compile-verified, matrix operand order checked by reading, not run on hardware
//
#include <hip/hip_runtime.h>
#include <stdint.h>

typedef __attribute__((ext_vector_type(16))) _Float16 v16h;
typedef __attribute__((ext_vector_type(8)))  _Float16 v8h;
typedef __attribute__((ext_vector_type(8)))  float    v8f;
typedef __attribute__((ext_vector_type(4)))  float    v4f;

#define S_LEN 4096
#define NB    2
#define NH    8
#define DK    64
#define DM    512
#define BAND  128
#define KT    17      // 16-wide key tiles covering [q0-128, q0+143]
#define PCW   288     // band-compact P row width (272 real + 16 pad), halfs

// ---- per-lane operand packing per CDNA5 ISA 16-bit A/B layout:
// lane half-index 0..7  -> K = kbase+0..7      (kbase = (lane>>4)*8)
// lane half-index 8..15 -> K = 16+kbase+0..7
__device__ __forceinline__ v16h load_h16(const _Float16* p) {
  v8h c1 = *(const v8h*)(p);
  v8h c2 = *(const v8h*)(p + 16);
  v16h r;
#pragma unroll
  for (int t = 0; t < 8; ++t) { r[t] = c1[t]; r[t + 8] = c2[t]; }
  return r;
}

__device__ __forceinline__ v16h load_f32_cvt_h16(const float* p) {
  v4f f0 = *(const v4f*)(p);
  v4f f1 = *(const v4f*)(p + 4);
  v4f f2 = *(const v4f*)(p + 16);
  v4f f3 = *(const v4f*)(p + 20);
  v16h r;
#pragma unroll
  for (int t = 0; t < 4; ++t) {
    r[t]      = (_Float16)f0[t];
    r[t + 4]  = (_Float16)f1[t];
    r[t + 8]  = (_Float16)f2[t];
    r[t + 12] = (_Float16)f3[t];
  }
  return r;
}

__device__ __forceinline__ v8f wmma_f16(v16h a, v16h b, v8f c) {
  // D = A(16x32 f16) * B(32x16 f16) + C(16x16 f32)
  return __builtin_amdgcn_wmma_f32_16x16x32_f16(false, a, false, b, (short)0, c,
                                                false, false);
}

// ---------------------------------------------------------------- convert
__global__ void cvt_f32_to_f16(const float* __restrict__ src,
                               _Float16* __restrict__ dst, int n) {
  int i = blockIdx.x * blockDim.x + threadIdx.x;
  const int stride = gridDim.x * blockDim.x;
  for (; i < n; i += stride) dst[i] = (_Float16)src[i];
}

// ---------------------------------------------------------------- GEMM: out = A @ W^T + bias
// A: [M,K] (f32 or f16, row-major), W: [N,K] f16 row-major, bias f32[N].
// One block = 4 waves sharing one 16-row stripe and 64 consecutive W columns;
// per k-step the block stages W[64][32] (4 KB) into LDS with
// GLOBAL_LOAD_ASYNC_TO_LDS_B128 (ASYNCcnt protocol), then each wave pulls its
// B operand from LDS (4-way reuse).
// OUT_MODE: 0 = f32 [M,N], 1 = f16 [M,N], 2 = f16 Vt[bh][d][s] (fused V transpose)
template <bool A_F32, int OUT_MODE>
__global__ void gemm_bias_wmma(const void* __restrict__ Av,
                               const _Float16* __restrict__ W,
                               const float* __restrict__ bias,
                               void* __restrict__ Ov, int M, int N, int K) {
  __shared__ _Float16 lw[64 * 32];   // W stage: 64 cols x 32 k, 4 KB

  const int tid  = threadIdx.x;
  const int lane = tid & 31;
  const int wave = tid >> 5;
  const int tileBase = blockIdx.x * 4;          // 4 waves = 4 N-tiles, same M-tile
  const int tilesN = N >> 4;                    // 32; divisible by 4
  const int tM = tileBase / tilesN;
  const int tNb = tileBase % tilesN;
  const int m16 = lane & 15;
  const int kb8 = (lane >> 4) * 8;
  const int rowA  = tM * 16 + m16;              // A row for this lane
  const int col   = (tNb + wave) * 16 + m16;    // B/C column for this lane
  const int cbase = tNb * 16;                   // first W column staged by block

  const float*    A32 = (const float*)Av;
  const _Float16* A16 = (const _Float16*)Av;

  v8f acc = {};
  for (int k0 = 0; k0 < K; k0 += 32) {
    // ---- cooperative stage of W[cbase..cbase+63][k0..k0+31] into LDS
#pragma unroll
    for (int cc = 0; cc < 2; ++cc) {
      const int ci = tid + cc * 128;            // 256 chunks of 8 halfs (16 B)
      const int wr = ci >> 2;                   // W column 0..63
      const int wk = (ci & 3) * 8;              // k offset 0/8/16/24
      const _Float16* gp = W + (size_t)(cbase + wr) * K + k0 + wk;
      _Float16* lp = lw + ci * 8;
#if defined(__gfx1250__)
      const unsigned lds_off = (unsigned)(uintptr_t)lp;   // addr[31:0] = LDS offset
      asm volatile("global_load_async_to_lds_b128 %0, %1, off"
                   :: "v"(lds_off), "v"((unsigned long long)(uintptr_t)gp)
                   : "memory");
#else
      *(v8h*)lp = *(const v8h*)gp;
#endif
    }
#if defined(__gfx1250__)
    asm volatile("s_wait_asynccnt 0x0" ::: "memory");
#endif
    __syncthreads();

    v16h a;
    if constexpr (A_F32) {
      a = load_f32_cvt_h16(A32 + (size_t)rowA * K + k0 + kb8);
      __builtin_prefetch((const void*)(A32 + (size_t)rowA * K + k0 + 32), 0, 0);
    } else {
      a = load_h16(A16 + (size_t)rowA * K + k0 + kb8);
      __builtin_prefetch((const void*)(A16 + (size_t)rowA * K + k0 + 32), 0, 0);
    }
    const v16h b = load_h16(lw + (wave * 16 + m16) * 32 + kb8);   // ds_load
    acc = wmma_f16(a, b, acc);
    __syncthreads();    // protect lw before next stage overwrites
  }

  const float bn = bias[col];
  const int rbase = tM * 16 + 8 * (lane >> 4);  // C row = v + 8*(lane>>4)
  if constexpr (OUT_MODE == 0) {                // f32 row-major (final output)
    float* O = (float*)Ov;
#pragma unroll
    for (int v = 0; v < 8; ++v)
      O[(size_t)(rbase + v) * N + col] = acc[v] + bn;
  } else if constexpr (OUT_MODE == 1) {         // f16 row-major (Q/K)
    _Float16* O = (_Float16*)Ov;
#pragma unroll
    for (int v = 0; v < 8; ++v)
      O[(size_t)(rbase + v) * N + col] = (_Float16)(acc[v] + bn);
  } else {                                      // f16 Vt[bh][d][s] (V, transposed)
    _Float16* O = (_Float16*)Ov;
    const int h = col >> 6, d = col & (DK - 1);
#pragma unroll
    for (int v = 0; v < 8; ++v) {
      const int row = rbase + v;
      const int b = row >> 12, s = row & (S_LEN - 1);
      O[((size_t)(b * NH + h) * DK + d) * S_LEN + s] = (_Float16)(acc[v] + bn);
    }
  }
}

// ---------------------------------------------------------------- banded scores + softmax
// one wave = one (b,h,16-query block); writes the FULL attn rows it owns
// (band probabilities + cooperative zero-fill of the complement) and the
// band-compact P (f16, [bh*S + i]*PCW + kk), kk -> key j = q0-128+kk.
__global__ void attn_scores(const _Float16* __restrict__ Qh,
                            const _Float16* __restrict__ Kh,
                            float* __restrict__ attn,
                            _Float16* __restrict__ Pc) {
  const int lane = threadIdx.x & 31;
  const int w = blockIdx.x * (blockDim.x >> 5) + (threadIdx.x >> 5);
  const int qblk = w & 255;
  const int bh = w >> 8;
  const int b = bh >> 3, h = bh & 7;
  const int q0 = qblk * 16;
  const int m16 = lane & 15;
  const int kb8 = (lane >> 4) * 8;
  const float scale = 0.125f;   // 1/sqrt(64)

  // preload Q A-tiles: rows q0..q0+15, d in [0,32) and [32,64)
  const _Float16* qrow = Qh + ((size_t)(b * S_LEN + q0 + m16)) * DM + h * DK;
  const v16h a0 = load_h16(qrow + 0  + kb8);
  const v16h a1 = load_h16(qrow + 32 + kb8);

  float mrow[8], lrow[8];
#pragma unroll
  for (int v = 0; v < 8; ++v) { mrow[v] = -1e30f; lrow[v] = 0.0f; }

  // ---- pass 1: per-lane online softmax stats
  for (int kt = 0; kt < KT; ++kt) {
    const int j = q0 - BAND + kt * 16 + m16;
    const bool jv = (j >= 0) && (j < S_LEN);
    v16h b0 = {}, b1 = {};
    if (jv) {
      const _Float16* krow = Kh + ((size_t)(b * S_LEN + j)) * DM + h * DK;
      b0 = load_h16(krow + 0  + kb8);
      b1 = load_h16(krow + 32 + kb8);
    }
    v8f c = {};
    c = wmma_f16(a0, b0, c);
    c = wmma_f16(a1, b1, c);
#pragma unroll
    for (int v = 0; v < 8; ++v) {
      const int i = q0 + v + 8 * (lane >> 4);
      const bool band = jv && (i - j <= BAND) && (j - i <= BAND);
      const float s = band ? c[v] * scale : -1e9f;
      const float nm = fmaxf(mrow[v], s);
      lrow[v] = lrow[v] * __expf(mrow[v] - nm) + __expf(s - nm);
      mrow[v] = nm;
    }
  }
  // merge (m,l) across the 16 lanes sharing each row
#pragma unroll
  for (int off = 1; off < 16; off <<= 1) {
#pragma unroll
    for (int v = 0; v < 8; ++v) {
      const float mo = __shfl_xor(mrow[v], off, 32);
      const float lo = __shfl_xor(lrow[v], off, 32);
      const float nm = fmaxf(mrow[v], mo);
      lrow[v] = lrow[v] * __expf(mrow[v] - nm) + lo * __expf(mo - nm);
      mrow[v] = nm;
    }
  }
#pragma unroll
  for (int v = 0; v < 8; ++v) lrow[v] = 1.0f / lrow[v];  // diagonal guarantees l>0

  // ---- pass 2: probabilities
  for (int kt = 0; kt < KT; ++kt) {
    const int j = q0 - BAND + kt * 16 + m16;
    const bool jv = (j >= 0) && (j < S_LEN);
    v16h b0 = {}, b1 = {};
    if (jv) {
      const _Float16* krow = Kh + ((size_t)(b * S_LEN + j)) * DM + h * DK;
      b0 = load_h16(krow + 0  + kb8);
      b1 = load_h16(krow + 32 + kb8);
    }
    v8f c = {};
    c = wmma_f16(a0, b0, c);
    c = wmma_f16(a1, b1, c);
#pragma unroll
    for (int v = 0; v < 8; ++v) {
      const int i = q0 + v + 8 * (lane >> 4);
      const bool band = jv && (i - j <= BAND) && (j - i <= BAND);
      const float s = band ? c[v] * scale : -1e9f;
      const float p = __expf(s - mrow[v]) * lrow[v];    // ==0 when masked
      if (jv) attn[((size_t)bh * S_LEN + i) * S_LEN + j] = p;
      Pc[((size_t)bh * S_LEN + i) * PCW + kt * 16 + m16] = (_Float16)p;
    }
  }
  // zero the pad columns kk = 272..287 for all 16 rows
  {
    const int i = q0 + m16;
#pragma unroll
    for (int t = 0; t < 8; ++t)
      Pc[((size_t)bh * S_LEN + i) * PCW + 272 + kb8 + t] = (_Float16)0.0f;
  }
  // ---- cooperative zero-fill of the out-of-band attn columns (fused, so the
  //      1.07 GB attn region is written exactly once per call)
  {
    const int jlo = (q0 - BAND > 0) ? (q0 - BAND) : 0;          // 64B-aligned
    const int jhi = (q0 + BAND + 16 < S_LEN) ? (q0 + BAND + 16) : S_LEN;
    const v4f z = {};
    for (int r = 0; r < 16; ++r) {
      float* rp = attn + ((size_t)bh * S_LEN + q0 + r) * S_LEN;
      for (int c = lane * 4; c < jlo; c += 128) *(v4f*)(rp + c) = z;
      for (int c = jhi + lane * 4; c < S_LEN; c += 128) *(v4f*)(rp + c) = z;
    }
  }
}

// ---------------------------------------------------------------- PV: ctx = P @ V (banded)
__global__ void attn_pv(const _Float16* __restrict__ Pc,
                        const _Float16* __restrict__ Vt,
                        _Float16* __restrict__ ctx) {
  const int lane = threadIdx.x & 31;
  const int w = blockIdx.x * (blockDim.x >> 5) + (threadIdx.x >> 5);
  const int qblk = w & 255;
  const int bh = w >> 8;
  const int b = bh >> 3, h = bh & 7;
  const int q0 = qblk * 16;
  const int m16 = lane & 15;
  const int kb8 = (lane >> 4) * 8;

  const _Float16* prow = Pc + ((size_t)bh * S_LEN + q0 + m16) * PCW;
  const _Float16* vtb  = Vt + (size_t)bh * DK * S_LEN;

  v8f acc[4] = {{}, {}, {}, {}};
  for (int kk0 = 0; kk0 < PCW; kk0 += 32) {
    const v16h a = load_h16(prow + kk0 + kb8);   // P, contiguous along kk
    const int j1 = q0 - BAND + kk0 + kb8;        // key index of chunk 1 (8-aligned)
    const int j2 = j1 + 16;                      // key index of chunk 2
#pragma unroll
    for (int dt = 0; dt < 4; ++dt) {
      const _Float16* vrow = vtb + (size_t)(dt * 16 + m16) * S_LEN;
      v16h bm = {};
      if (j1 >= 0 && j1 < S_LEN) {
        v8h c = *(const v8h*)(vrow + j1);
#pragma unroll
        for (int t = 0; t < 8; ++t) bm[t] = c[t];
      }
      if (j2 >= 0 && j2 < S_LEN) {
        v8h c = *(const v8h*)(vrow + j2);
#pragma unroll
        for (int t = 0; t < 8; ++t) bm[8 + t] = c[t];
      }
      acc[dt] = wmma_f16(a, bm, acc[dt]);
    }
  }
#pragma unroll
  for (int dt = 0; dt < 4; ++dt) {
#pragma unroll
    for (int v = 0; v < 8; ++v) {
      const int i = q0 + v + 8 * (lane >> 4);
      ctx[((size_t)(b * S_LEN + i)) * DM + h * DK + dt * 16 + m16] =
          (_Float16)acc[dt][v];
    }
  }
}

// ---------------------------------------------------------------- host side
extern "C" void kernel_launch(void* const* d_in, const int* in_sizes, int n_in,
                              void* d_out, int out_size, void* d_ws, size_t ws_size,
                              hipStream_t stream) {
  const float* q  = (const float*)d_in[0];
  const float* k  = (const float*)d_in[1];
  const float* vv = (const float*)d_in[2];
  const float* Wq = (const float*)d_in[3];
  const float* bq = (const float*)d_in[4];
  const float* Wk = (const float*)d_in[5];
  const float* bk = (const float*)d_in[6];
  const float* Wv = (const float*)d_in[7];
  const float* bv = (const float*)d_in[8];
  const float* Wo = (const float*)d_in[9];
  const float* bo = (const float*)d_in[10];

  float* out  = (float*)d_out;
  float* attn = out + (size_t)NB * S_LEN * DM;   // second tuple element

  char* ws = (char*)d_ws;
  size_t off = 0;
  auto carve = [&](size_t bytes) -> char* {
    char* p = ws + off;
    off += (bytes + 255) & ~(size_t)255;
    return p;
  };
  const size_t WSZ = (size_t)DM * DM * sizeof(_Float16);         // 512 KB
  const size_t XSZ = (size_t)NB * S_LEN * DM * sizeof(_Float16); // 8.39 MB
  _Float16* wq16 = (_Float16*)carve(WSZ);
  _Float16* wk16 = (_Float16*)carve(WSZ);
  _Float16* wv16 = (_Float16*)carve(WSZ);
  _Float16* wo16 = (_Float16*)carve(WSZ);
  _Float16* Qh   = (_Float16*)carve(XSZ);
  _Float16* Kh   = (_Float16*)carve(XSZ);
  _Float16* Vth  = (_Float16*)carve(XSZ);   // V written directly transposed
  _Float16* Pc   = (_Float16*)carve((size_t)NB * NH * S_LEN * PCW * sizeof(_Float16));
  _Float16* ctx  = (_Float16*)carve(XSZ);
  (void)ws_size; (void)in_sizes; (void)n_in; (void)out_size;

  const int WN = DM * DM;
  cvt_f32_to_f16<<<256, 256, 0, stream>>>(Wq, wq16, WN);
  cvt_f32_to_f16<<<256, 256, 0, stream>>>(Wk, wk16, WN);
  cvt_f32_to_f16<<<256, 256, 0, stream>>>(Wv, wv16, WN);
  cvt_f32_to_f16<<<256, 256, 0, stream>>>(Wo, wo16, WN);

  const int M = NB * S_LEN;                          // 8192 rows
  const int gemmBlocks = (M / 16) * (DM / 16) / 4;   // 4 waves/block, exact
  gemm_bias_wmma<true, 1><<<gemmBlocks, 128, 0, stream>>>(q,  wq16, bq, Qh,  M, DM, DM);
  gemm_bias_wmma<true, 1><<<gemmBlocks, 128, 0, stream>>>(k,  wk16, bk, Kh,  M, DM, DM);
  gemm_bias_wmma<true, 2><<<gemmBlocks, 128, 0, stream>>>(vv, wv16, bv, Vth, M, DM, DM);

  const int attnWaves = NB * NH * (S_LEN / 16);      // 4096
  attn_scores<<<attnWaves / 4, 128, 0, stream>>>(Qh, Kh, attn, Pc);
  attn_pv    <<<attnWaves / 4, 128, 0, stream>>>(Pc, Vth, ctx);

  gemm_bias_wmma<false, 0><<<gemmBlocks, 128, 0, stream>>>(ctx, wo16, bo, out, M, DM, DM);
}
